// VGAE_15066745274948
// MI455X (gfx1250) — matrix operand order repaired and checked by reading, not compiled, not run on hardware
//
#include <hip/hip_runtime.h>
#include <hip/hip_bf16.h>

#define NN 50000
#define NE 800000
#define DIN 256
#define DEMB1 128
#define DEMB2 64
#define DL1 64

typedef __attribute__((ext_vector_type(16))) __bf16 v16bf;
typedef __attribute__((ext_vector_type(8)))  float  v8f;

// ---------------- activation ----------------
__device__ __forceinline__ float act_apply(float v, int act) {
  switch (act) {
    case 1: return v > 0.f ? v : 0.2f * v;      // leaky_relu(0.2)
    case 2: return tanhf(v);                    // tanh
    case 3: return expf(tanhf(v));              // z_std = exp(tanh(.))
    default: return v;
  }
}

// ---------------- WMMA GEMM: D = act(X @ W^T + bias) ----------------
// X: [nrows, K] f32 row-major, W: [F, K] f32 row-major, D: [nrows, F] f32.
// One wave computes a 16x64 tile (4 WMMA accumulators): A loaded once per
// K-step (bf16), reused across 4 column sub-tiles; f32 accumulate.
// Requires: nrows % 16 == 0, K % 32 == 0, F % 64 == 0.
__global__ __launch_bounds__(256)
void gemm_bf16_wmma(const float* __restrict__ X, const float* __restrict__ W,
                    const float* __restrict__ bias, float* __restrict__ D,
                    int nrows, int K, int F, int act) {
  int wavesPerBlock = blockDim.x >> 5;
  int wid = blockIdx.x * wavesPerBlock + (threadIdx.x >> 5);
  int ctiles = F >> 6;                      // 64-wide column tiles
  int total = (nrows >> 4) * ctiles;
  if (wid >= total) return;                 // wave-uniform: EXEC stays all-ones
  int mt = wid / ctiles;
  int ct = wid - mt * ctiles;
  int lane = threadIdx.x & 31;
  int half = lane >> 4;                     // 0: lanes 0-15, 1: lanes 16-31
  int l15  = lane & 15;
  int row  = (mt << 4) + l15;               // A-matrix row for this lane
  int colb = (ct << 6) + l15;               // first of 4 B/D columns
  const float* __restrict__ xrow = X + (size_t)row * K;
  const float* __restrict__ wr0 = W + (size_t)(colb +  0) * K;
  const float* __restrict__ wr1 = W + (size_t)(colb + 16) * K;
  const float* __restrict__ wr2 = W + (size_t)(colb + 32) * K;
  const float* __restrict__ wr3 = W + (size_t)(colb + 48) * K;
  v8f c0 = {}, c1 = {}, c2 = {}, c3 = {};
  for (int k0 = 0; k0 < K; k0 += 32) {
    // A (16x32 bf16): lane half 0 -> K = k0+{0..15}; half 1 -> k0+{16..31}
    const float* xa = xrow + k0 + half * 16;
    v16bf a;
#pragma unroll
    for (int j = 0; j < 8; ++j) {
      a[j]     = (__bf16)xa[j];
      a[j + 8] = (__bf16)xa[j + 8];
    }
    // B (32x16 bf16): lane half 0 -> K = k0+{0..15}; half 1 -> k0+{16..31}
    int kb = k0 + half * 16;
    v16bf b0, b1, b2, b3;
#pragma unroll
    for (int j = 0; j < 16; ++j) b0[j] = (__bf16)wr0[kb + j];
#pragma unroll
    for (int j = 0; j < 16; ++j) b1[j] = (__bf16)wr1[kb + j];
#pragma unroll
    for (int j = 0; j < 16; ++j) b2[j] = (__bf16)wr2[kb + j];
#pragma unroll
    for (int j = 0; j < 16; ++j) b3[j] = (__bf16)wr3[kb + j];
    c0 = __builtin_amdgcn_wmma_f32_16x16x32_bf16(false, a, false, b0, (short)0, c0, false, false);
    c1 = __builtin_amdgcn_wmma_f32_16x16x32_bf16(false, a, false, b1, (short)0, c1, false, false);
    c2 = __builtin_amdgcn_wmma_f32_16x16x32_bf16(false, a, false, b2, (short)0, c2, false, false);
    c3 = __builtin_amdgcn_wmma_f32_16x16x32_bf16(false, a, false, b3, (short)0, c3, false, false);
  }
  int mbase = (mt << 4) + half * 8;         // C/D: VGPR i -> M = mbase + i
  v8f accs[4] = {c0, c1, c2, c3};
#pragma unroll
  for (int s = 0; s < 4; ++s) {
    int col = colb + 16 * s;
    float bv = bias ? bias[col] : 0.f;
#pragma unroll
    for (int i = 0; i < 8; ++i) {
      D[(size_t)(mbase + i) * F + col] = act_apply(accs[s][i] + bv, act);
    }
  }
}

// ---------------- utility kernels ----------------
__global__ void fill_f32(float* __restrict__ p, float v, long long n) {
  long long i = blockIdx.x * (long long)blockDim.x + threadIdx.x;
  if (i < n) p[i] = v;
}

// dst[row[e]*F + f] += src[col[e]*F + f], F = 1<<logF
__global__ void scatter_add_pow2(const float* __restrict__ src,
                                 const int* __restrict__ rw,
                                 const int* __restrict__ cl,
                                 float* __restrict__ dst, int E, int logF) {
  long long gid = blockIdx.x * (long long)blockDim.x + threadIdx.x;
  long long total = (long long)E << logF;
  if (gid >= total) return;
  int e = (int)(gid >> logF);
  int f = (int)(gid & ((1ll << logF) - 1));
  atomicAdd(dst + ((size_t)rw[e] << logF) + f, src[((size_t)cl[e] << logF) + f]);
}

__global__ void ew_add_act(float* __restrict__ dst, const float* __restrict__ a,
                           const float* __restrict__ b, long long n, int act) {
  long long i = blockIdx.x * (long long)blockDim.x + threadIdx.x;
  if (i < n) dst[i] = act_apply(a[i] + b[i], act);
}

__global__ void ew_add_tanh2(float* __restrict__ d1, float* __restrict__ d2,
                             const float* __restrict__ a, const float* __restrict__ b,
                             long long n) {
  long long i = blockIdx.x * (long long)blockDim.x + threadIdx.x;
  if (i < n) { float v = tanhf(a[i] + b[i]); d1[i] = v; d2[i] = v; }
}

__device__ __forceinline__ float wave_sum(float v) {
#pragma unroll
  for (int o = 16; o > 0; o >>= 1) v += __shfl_xor(v, o, 32);
  return v;
}

// inv[n] = 1 / max(||z[n]||, 1e-8) ; one wave per node (EMB2=64 -> 2 per lane)
__global__ void node_inv_norm(const float* __restrict__ z, float* __restrict__ inv, int n) {
  int w = blockIdx.x * (blockDim.x >> 5) + (threadIdx.x >> 5);
  if (w >= n) return;
  int lane = threadIdx.x & 31;
  const float* zr = z + (size_t)w * DEMB2;
  float a0 = zr[lane], a1 = zr[lane + 32];
  float s = wave_sum(a0 * a0 + a1 * a1);
  if (lane == 0) inv[w] = 1.f / fmaxf(sqrtf(s), 1e-8f);
}

// w_mu[e] = dot(z[row],z[col]) * inv[row] * inv[col] ; one wave per edge
__global__ void edge_cosine(const float* __restrict__ z, const float* __restrict__ inv,
                            const int* __restrict__ rw, const int* __restrict__ cl,
                            float* __restrict__ out, int E) {
  int w = blockIdx.x * (blockDim.x >> 5) + (threadIdx.x >> 5);
  if (w >= E) return;
  int lane = threadIdx.x & 31;
  int r = rw[w], c = cl[w];
  const float* za = z + (size_t)r * DEMB2;
  const float* zb = z + (size_t)c * DEMB2;
  float s = wave_sum(za[lane] * zb[lane] + za[lane + 32] * zb[lane + 32]);
  if (lane == 0) out[w] = s * inv[r] * inv[c];
}

// pr[n] = dot(g1[n], w_rel_g2) ; rootp[n] = dot(g1[n], w_root_g2) + b_g2
__global__ void gate_dots(const float* __restrict__ g1, const float* __restrict__ wrel,
                          const float* __restrict__ wroot, const float* __restrict__ bg2,
                          float* __restrict__ pr, float* __restrict__ rootp, int n) {
  int w = blockIdx.x * (blockDim.x >> 5) + (threadIdx.x >> 5);
  if (w >= n) return;
  int lane = threadIdx.x & 31;
  const float* g = g1 + (size_t)w * DL1;
  float x0 = g[lane], x1 = g[lane + 32];
  float sr = wave_sum(x0 * wrel[lane] + x1 * wrel[lane + 32]);
  float so = wave_sum(x0 * wroot[lane] + x1 * wroot[lane + 32]);
  if (lane == 0) { pr[w] = sr; rootp[w] = so + bg2[0]; }
}

// ---- deterministic softmax-over-N reductions ----
__global__ void reduce_max_p1(const float* __restrict__ g, float* __restrict__ part, int n) {
  __shared__ float sm[256];
  float m = -3.0e38f;
  for (int i = blockIdx.x * blockDim.x + threadIdx.x; i < n; i += gridDim.x * blockDim.x)
    m = fmaxf(m, g[i]);
  sm[threadIdx.x] = m; __syncthreads();
  for (int s = 128; s > 0; s >>= 1) {
    if (threadIdx.x < s) sm[threadIdx.x] = fmaxf(sm[threadIdx.x], sm[threadIdx.x + s]);
    __syncthreads();
  }
  if (threadIdx.x == 0) part[blockIdx.x] = sm[0];
}
__global__ void reduce_max_p2(const float* __restrict__ part, float* __restrict__ out, int n) {
  __shared__ float sm[256];
  float m = -3.0e38f;
  for (int i = threadIdx.x; i < n; i += blockDim.x) m = fmaxf(m, part[i]);
  sm[threadIdx.x] = m; __syncthreads();
  for (int s = 128; s > 0; s >>= 1) {
    if (threadIdx.x < s) sm[threadIdx.x] = fmaxf(sm[threadIdx.x], sm[threadIdx.x + s]);
    __syncthreads();
  }
  if (threadIdx.x == 0) out[0] = sm[0];
}
__global__ void reduce_sumexp_p1(const float* __restrict__ g, const float* __restrict__ mptr,
                                 float* __restrict__ part, int n) {
  __shared__ float sm[256];
  float m = mptr[0], acc = 0.f;
  for (int i = blockIdx.x * blockDim.x + threadIdx.x; i < n; i += gridDim.x * blockDim.x)
    acc += expf(g[i] - m);
  sm[threadIdx.x] = acc; __syncthreads();
  for (int s = 128; s > 0; s >>= 1) {
    if (threadIdx.x < s) sm[threadIdx.x] += sm[threadIdx.x + s];
    __syncthreads();
  }
  if (threadIdx.x == 0) part[blockIdx.x] = sm[0];
}
__global__ void reduce_sum_p2(const float* __restrict__ part, float* __restrict__ out, int n) {
  __shared__ float sm[256];
  float acc = 0.f;
  for (int i = threadIdx.x; i < n; i += blockDim.x) acc += part[i];
  sm[threadIdx.x] = acc; __syncthreads();
  for (int s = 128; s > 0; s >>= 1) {
    if (threadIdx.x < s) sm[threadIdx.x] += sm[threadIdx.x + s];
    __syncthreads();
  }
  if (threadIdx.x == 0) out[0] = sm[0];
}
// pooled[f] = sum_n softmax(gate)[n] * z[n][f] ; one block per feature f
__global__ void pooled_kernel(const float* __restrict__ gate, const float* __restrict__ z,
                              const float* __restrict__ mptr, const float* __restrict__ sptr,
                              float* __restrict__ pooled, int n) {
  __shared__ float sm[256];
  int f = blockIdx.x;
  float m = mptr[0], S = sptr[0], acc = 0.f;
  for (int i = threadIdx.x; i < n; i += blockDim.x)
    acc += expf(gate[i] - m) * z[(size_t)i * DEMB2 + f];
  sm[threadIdx.x] = acc; __syncthreads();
  for (int s = 128; s > 0; s >>= 1) {
    if (threadIdx.x < s) sm[threadIdx.x] += sm[threadIdx.x + s];
    __syncthreads();
  }
  if (threadIdx.x == 0) pooled[f] = sm[0] / S;
}

// classifier: y = softmax(leaky(pooled@W1^T+b1)@W2^T+b2); also w_std_out
__global__ void classifier_kernel(const float* __restrict__ pooled,
                                  const float* __restrict__ w1, const float* __restrict__ b1,
                                  const float* __restrict__ w2, const float* __restrict__ b2,
                                  const float* __restrict__ log_std, float* __restrict__ out) {
  __shared__ float y1[64];
  __shared__ float y2[2];
  int t = threadIdx.x;   // blockDim = 64
  float acc = b1[t];
  for (int k = 0; k < 64; ++k) acc += pooled[k] * w1[t * 64 + k];
  y1[t] = acc > 0.f ? acc : 0.2f * acc;
  __syncthreads();
  if (t < 2) {
    float a2 = b2[t];
    for (int j = 0; j < 64; ++j) a2 += y1[j] * w2[t * 64 + j];
    y2[t] = a2;
  }
  __syncthreads();
  if (t == 0) {
    float m = fmaxf(y2[0], y2[1]);
    float e0 = expf(y2[0] - m), e1 = expf(y2[1] - m);
    out[0] = e0 / (e0 + e1);
    out[1] = e1 / (e0 + e1);
    out[800002] = expf(log_std[0]);   // w_std_out
  }
}

// ---------------- host orchestration ----------------
static inline int cdivll(long long a, long long b) { return (int)((a + b - 1) / b); }

extern "C" void kernel_launch(void* const* d_in, const int* in_sizes, int n_in,
                              void* d_out, int out_size, void* d_ws, size_t ws_size,
                              hipStream_t stream) {
  (void)in_sizes; (void)n_in; (void)out_size; (void)ws_size;
  const float* x        = (const float*)d_in[0];
  const int*   rowI     = (const int*)  d_in[1];
  const int*   colI     = (const int*)  d_in[2];
  const float* w_rel1   = (const float*)d_in[3];
  const float* w_root1  = (const float*)d_in[4];
  const float* b1       = (const float*)d_in[5];
  const float* w_rel_mu = (const float*)d_in[6];
  const float* w_root_mu= (const float*)d_in[7];
  const float* b_mu     = (const float*)d_in[8];
  const float* w_std    = (const float*)d_in[9];
  const float* b_std    = (const float*)d_in[10];
  const float* w_rel_g1 = (const float*)d_in[11];
  const float* w_root_g1= (const float*)d_in[12];
  const float* b_g1     = (const float*)d_in[13];
  const float* w_rel_g2 = (const float*)d_in[14];
  const float* w_root_g2= (const float*)d_in[15];
  const float* b_g2     = (const float*)d_in[16];
  const float* w_cls1   = (const float*)d_in[17];
  const float* b_cls1   = (const float*)d_in[18];
  const float* w_cls2   = (const float*)d_in[19];
  const float* b_cls2   = (const float*)d_in[20];
  const float* log_std  = (const float*)d_in[21];

  float* out = (float*)d_out;
  // output tuple layout (flat): y[2], w_mu[E], w_std[1], z, z_mu, z_std
  const size_t OUT_WMU  = 2;
  const size_t OUT_Z    = 800003;
  const size_t OUT_ZMU  = 4000003;
  const size_t OUT_ZSTD = 7200003;

  float* ws   = (float*)d_ws;
  float* buf0 = ws;                           // [N,128]
  float* buf1 = ws + (size_t)NN * 128;        // [N,128]
  float* buf2 = ws + (size_t)NN * 256;        // [N,128]
  float* tail = ws + (size_t)NN * 384;        // small tail
  float* part1  = tail;                       // 256
  float* part2  = tail + 256;                 // 256
  float* scal   = tail + 512;                 // [0]=max, [1]=sumexp
  float* pooled = tail + 516;                 // 64

  // ---- layer 1: p1 = x@w_rel1^T, q1 = x@w_root1^T + b1 ----
  int waves1 = (NN / 16) * (DEMB1 / 64);      // 16x64 tile per wave
  gemm_bf16_wmma<<<cdivll(waves1, 8), 256, 0, stream>>>(x, w_rel1, nullptr, buf0, NN, DIN, DEMB1, 0);
  gemm_bf16_wmma<<<cdivll(waves1, 8), 256, 0, stream>>>(x, w_root1, b1, buf1, NN, DIN, DEMB1, 0);
  // agg1 = A @ p1 (scatter-add over edges), h = leaky(agg1 + q1) -> buf2
  fill_f32<<<cdivll((long long)NN * 128, 256), 256, 0, stream>>>(buf2, 0.f, (long long)NN * 128);
  scatter_add_pow2<<<cdivll((long long)NE << 7, 256), 256, 0, stream>>>(buf0, rowI, colI, buf2, NE, 7);
  ew_add_act<<<cdivll((long long)NN * 128, 256), 256, 0, stream>>>(buf2, buf2, buf1, (long long)NN * 128, 1);

  // ---- mu/std layer from h (buf2) ----
  int waves2 = (NN / 16) * (DEMB2 / 64);      // F=64 -> one 64-wide tile
  float* p_mu = buf0;
  float* q_mu = buf0 + (size_t)NN * 64;
  gemm_bf16_wmma<<<cdivll(waves2, 8), 256, 0, stream>>>(buf2, w_rel_mu, nullptr, p_mu, NN, DEMB1, DEMB2, 0);
  gemm_bf16_wmma<<<cdivll(waves2, 8), 256, 0, stream>>>(buf2, w_root_mu, b_mu, q_mu, NN, DEMB1, DEMB2, 0);
  gemm_bf16_wmma<<<cdivll(waves2, 8), 256, 0, stream>>>(buf2, w_std, b_std, out + OUT_ZSTD, NN, DEMB1, DEMB2, 3);
  float* agg_mu = buf1;
  fill_f32<<<cdivll((long long)NN * 64, 256), 256, 0, stream>>>(agg_mu, 0.f, (long long)NN * 64);
  scatter_add_pow2<<<cdivll((long long)NE << 6, 256), 256, 0, stream>>>(p_mu, rowI, colI, agg_mu, NE, 6);
  // z = z_mu = tanh(agg_mu + q_mu), written to both output regions
  ew_add_tanh2<<<cdivll((long long)NN * 64, 256), 256, 0, stream>>>(out + OUT_Z, out + OUT_ZMU, agg_mu, q_mu, (long long)NN * 64);

  // ---- decode: per-edge cosine similarity ----
  float* z = out + OUT_Z;
  float* invn = buf2;    // h no longer needed
  node_inv_norm<<<cdivll(NN, 8), 256, 0, stream>>>(z, invn, NN);
  edge_cosine<<<cdivll(NE, 8), 256, 0, stream>>>(z, invn, rowI, colI, out + OUT_WMU, NE);

  // ---- gate layer g1 = leaky(GraphConv(z)) ----
  float* p_g1 = buf0;
  float* q_g1 = buf0 + (size_t)NN * 64;
  gemm_bf16_wmma<<<cdivll(waves2, 8), 256, 0, stream>>>(z, w_rel_g1, nullptr, p_g1, NN, DEMB2, DL1, 0);
  gemm_bf16_wmma<<<cdivll(waves2, 8), 256, 0, stream>>>(z, w_root_g1, b_g1, q_g1, NN, DEMB2, DL1, 0);
  float* agg_g1 = buf1;
  fill_f32<<<cdivll((long long)NN * 64, 256), 256, 0, stream>>>(agg_g1, 0.f, (long long)NN * 64);
  scatter_add_pow2<<<cdivll((long long)NE << 6, 256), 256, 0, stream>>>(p_g1, rowI, colI, agg_g1, NE, 6);
  float* g1buf = buf1 + (size_t)NN * 64;
  ew_add_act<<<cdivll((long long)NN * 64, 256), 256, 0, stream>>>(g1buf, agg_g1, q_g1, (long long)NN * 64, 1);

  // ---- gate = GraphConv(g1) -> [N,1], softmax over N, pooled = gate^T @ z ----
  float* pr       = buf2;
  float* rootp    = buf2 + (size_t)NN;
  float* agg_gate = buf2 + (size_t)2 * NN;
  float* gate     = buf2 + (size_t)3 * NN;
  gate_dots<<<cdivll(NN, 8), 256, 0, stream>>>(g1buf, w_rel_g2, w_root_g2, b_g2, pr, rootp, NN);
  fill_f32<<<cdivll(NN, 256), 256, 0, stream>>>(agg_gate, 0.f, (long long)NN);
  scatter_add_pow2<<<cdivll(NE, 256), 256, 0, stream>>>(pr, rowI, colI, agg_gate, NE, 0);
  ew_add_act<<<cdivll(NN, 256), 256, 0, stream>>>(gate, agg_gate, rootp, (long long)NN, 0);
  reduce_max_p1<<<256, 256, 0, stream>>>(gate, part1, NN);
  reduce_max_p2<<<1, 256, 0, stream>>>(part1, scal, 256);
  reduce_sumexp_p1<<<256, 256, 0, stream>>>(gate, scal, part2, NN);
  reduce_sum_p2<<<1, 256, 0, stream>>>(part2, scal + 1, 256);
  pooled_kernel<<<64, 256, 0, stream>>>(gate, z, scal, scal + 1, pooled, NN);

  // ---- classifier + w_std_out ----
  classifier_kernel<<<1, 64, 0, stream>>>(pooled, w_cls1, b_cls1, w_cls2, b_cls2, log_std, out);
}
